// EARLModel_44538810859711
// MI455X (gfx1250) — matrix operand order repaired and checked by reading, not compiled, not run on hardware
//
#include <hip/hip_runtime.h>
#include <hip/hip_bf16.h>
#include <math.h>

// ---------------- problem constants ----------------
#define BB   16
#define LL   256
#define SS   100
#define LA_  48
#define NA_  32
#define EE   64
#define DD   768
#define HH   12
#define NHID 1024
#define WIDX 102
#define SP   112   // S padded to 16
#define KS   128   // S padded to 32 (K dim)
#define KA   64    // LA padded to 32 (K dim)
#define EPS_ 1e-12f

// ---------------- bf16 helpers ----------------
typedef __attribute__((ext_vector_type(16))) __bf16 v16bf;
typedef __attribute__((ext_vector_type(8)))  float  v8f;

struct alignas(16) U4 { unsigned x, y, z, w; };
union FragBF { v16bf v; U4 q[2]; };

__device__ __forceinline__ unsigned short f2bf(float f) {
  unsigned u = __builtin_bit_cast(unsigned, f);
  u += 0x7FFFu + ((u >> 16) & 1u);          // round-to-nearest-even
  return (unsigned short)(u >> 16);
}

// =====================================================================
// Register-blocked batched GEMM:
//   C[M x N] = scale * A[M x K](bf16) * Bt[N x K]^T(bf16)
// One wave computes a BM x BN grid of 16x16 tiles (BM+BN fragment loads
// feed BM*BN WMMAs per k-step; 8 independent accumulation chains at
// (2,4) hide WMMA latency in-wave).
// epilogue: (+bias[n]) (*rowmask[m]); writes any of: Cf (f32), Cb (bf16),
// Ct (bf16 transposed [N x M]).
// A fragment (16x32 bf16): lane r=l&15,h=l>>4 : elems 0..7  -> K=k0+8h+j
//                                               elems 8..15 -> K=k0+16+8h+j
// B fragment from Bt:      same K mapping as A fragment
// C mapping: c[v] -> (M = m0+v+8h, N = n0+r)
// =====================================================================
template<int BM, int BN>
__global__ __launch_bounds__(32) void k_gemm_bf16(
    const unsigned short* __restrict__ A,  long long sA,  int ldA,
    const unsigned short* __restrict__ Bt, long long sB,  int ldB,
    float* Cf,          long long sCf, int ldCf,
    unsigned short* Cb, long long sCb, int ldCb,
    unsigned short* Ct, long long sCt, int ldCt,
    const float* bias, const float* rmask, long long sMask,
    float scale, int K)
{
  const int nb = blockIdx.x, mb = blockIdx.y, b = blockIdx.z;
  const int lane = threadIdx.x;
  const int r = lane & 15, half = lane >> 4;
  const int aoff = half * 8;
  const int boff = half * 8;

  const unsigned short* Ab[BM];
#pragma unroll
  for (int i = 0; i < BM; ++i)
    Ab[i] = A + (long long)b * sA + (long long)((mb * BM + i) * 16 + r) * ldA;
  const unsigned short* Bb[BN];
#pragma unroll
  for (int j = 0; j < BN; ++j)
    Bb[j] = Bt + (long long)b * sB + (long long)((nb * BN + j) * 16 + r) * ldB;

  v8f acc[BM][BN];
#pragma unroll
  for (int i = 0; i < BM; ++i)
#pragma unroll
    for (int j = 0; j < BN; ++j)
      acc[i][j] = (v8f){0.f, 0.f, 0.f, 0.f, 0.f, 0.f, 0.f, 0.f};

  for (int k0 = 0; k0 < K; k0 += 32) {
    FragBF af[BM], bf[BN];
#pragma unroll
    for (int i = 0; i < BM; ++i) {
      af[i].q[0] = *(const U4*)(Ab[i] + k0 + aoff);
      af[i].q[1] = *(const U4*)(Ab[i] + k0 + 16 + aoff);
    }
#pragma unroll
    for (int j = 0; j < BN; ++j) {
      bf[j].q[0] = *(const U4*)(Bb[j] + k0 + boff);
      bf[j].q[1] = *(const U4*)(Bb[j] + k0 + boff + 16);
    }
    __builtin_prefetch(Bb[0] + k0 + 64, 0, 3);
#pragma unroll
    for (int i = 0; i < BM; ++i)
#pragma unroll
      for (int j = 0; j < BN; ++j)
        acc[i][j] = __builtin_amdgcn_wmma_f32_16x16x32_bf16(
            false, af[i].v, false, bf[j].v, (short)0, acc[i][j], false, false);
  }

#pragma unroll
  for (int i = 0; i < BM; ++i)
#pragma unroll
    for (int j = 0; j < BN; ++j)
#pragma unroll
      for (int v = 0; v < 8; ++v) {
        const int m = (mb * BM + i) * 16 + v + 8 * half;
        const int n = (nb * BN + j) * 16 + r;
        float x = acc[i][j][v] * scale;
        if (bias)  x += bias[n];
        if (rmask) x *= rmask[(long long)b * sMask + m];
        if (Cf) Cf[(long long)b * sCf + (long long)m * ldCf + n] = x;
        if (Cb) Cb[(long long)b * sCb + (long long)m * ldCb + n] = f2bf(x);
        if (Ct) Ct[(long long)b * sCt + (long long)n * ldCt + m] = f2bf(x);
      }
}

// =====================================================================
// Pack weights to bf16, transposed to [N x K] operand layout.
// WtaT[768 x 2304], W1ET[1024 x 2304], W1AT[1024 x 2304]
// latent order: [H_, arg, tokA, A_h2h, argTok, A_u2u] (blocks of D rows of W1)
//   W1E rows: k<768 -> k        ; k>=768 -> k+768   (blocks 0,2,3)
//   W1A rows: k<768 -> k+768    ; k>=768 -> k+2304  (blocks 1,4,5)
// =====================================================================
__global__ void k_pack_w(const float* __restrict__ Wta, const float* __restrict__ W1,
                         unsigned short* WtaT, unsigned short* W1ET, unsigned short* W1AT)
{
  const long long NW = 768LL * 2304;
  const long long N1 = 1024LL * 2304;
  long long i = (long long)blockIdx.x * blockDim.x + threadIdx.x;
  if (i < NW) {
    int n = (int)(i / 2304), k = (int)(i % 2304);
    WtaT[i] = f2bf(Wta[(long long)k * 768 + n]);
  } else if (i < NW + N1) {
    long long j = i - NW;
    int n = (int)(j / 2304), k = (int)(j % 2304);
    int row = (k < 768) ? k : (k + 768);
    W1ET[j] = f2bf(W1[(long long)row * 1024 + n]);
  } else if (i < NW + 2 * N1) {
    long long j = i - NW - N1;
    int n = (int)(j / 2304), k = (int)(j % 2304);
    int row = (k < 768) ? (k + 768) : (k + 2304);
    W1AT[j] = f2bf(W1[(long long)row * 1024 + n]);
  }
}

// =====================================================================
// Gather: sentT bf16 [B][D][KS] (masked), trig f32 [B][D],
//         EMt bf16 [B][E][KS], argT bf16 [B][D][KA], argWt bf16 [B][NA][KA]
// grid (B, 4 sections), block 256
// =====================================================================
__global__ void k_gather(const float* __restrict__ emb, const float* __restrict__ istrig,
                         const float* __restrict__ argw, const float* __restrict__ entmap,
                         const int* __restrict__ idxs, const int* __restrict__ blens,
                         unsigned short* sentT, unsigned short* EMt,
                         unsigned short* argT, unsigned short* argWt, float* trig)
{
  const int b = blockIdx.x, sec = blockIdx.y, tid = threadIdx.x;
  int nv = 0;
  for (int w = 0; w < WIDX; ++w) nv += (idxs[b * WIDX + w] > 0) ? 1 : 0;
  const int sls  = idxs[b * WIDX + (nv - 2)];
  const int blen = blens[b];

  if (sec == 0) {
    float den = 0.f;
    for (int s = 0; s < SS; ++s) den += istrig[b * SS + s];
    for (int d = tid; d < DD; d += 256) {
      float tacc = 0.f;
      for (int s = 0; s < KS; ++s) {
        float v = 0.f;
        if (s < SS && s < nv - 2)
          v = emb[((long long)(b * LL + idxs[b * WIDX + s])) * DD + d];
        sentT[((long long)(b * DD + d)) * KS + s] = f2bf(v);
        if (s < SS) tacc += v * istrig[b * SS + s];
      }
      trig[b * DD + d] = tacc / den;
    }
  } else if (sec == 1) {
    for (int i = tid; i < EE * KS; i += 256) {
      int e = i >> 7, s = i & 127;
      float v = (s < SS) ? entmap[((long long)(b * SS + s)) * EE + e] : 0.f;
      EMt[(long long)b * (EE * KS) + i] = f2bf(v);
    }
  } else if (sec == 2) {
    for (int i = tid; i < DD * KA; i += 256) {
      int d = i >> 6, j = i & 63;
      int pos = sls + 1 + j;
      float v = 0.f;
      if (j < LA_ && pos < blen - 1) {
        int p = pos > (LL - 1) ? (LL - 1) : pos;
        v = emb[((long long)(b * LL + p)) * DD + d];
      }
      argT[(long long)b * (DD * KA) + i] = f2bf(v);
    }
  } else {
    for (int i = tid; i < NA_ * KA; i += 256) {
      int a = i >> 6, l = i & 63;
      float v = (l < LA_) ? argw[((long long)(b * LA_ + l)) * NA_ + a] : 0.f;
      argWt[(long long)b * (NA_ * KA) + i] = f2bf(v);
    }
  }
}

// =====================================================================
// Build ent_cat bf16 [B][E][2304] = [entity0, trig, entity0*trig]
// and nonempty[b][e] = (sum_d |entity0|) > 0.   grid B*E, block 256
// =====================================================================
__global__ void k_entcat(const float* __restrict__ entity0, const float* __restrict__ trig,
                         unsigned short* entcat, float* nonempty)
{
  const int be = blockIdx.x, b = be >> 6, tid = threadIdx.x;
  const float* row = entity0 + (long long)be * DD;
  __shared__ float red[256];
  float p = 0.f;
  for (int d = tid; d < DD; d += 256) p += fabsf(row[d]);
  red[tid] = p; __syncthreads();
  for (int o = 128; o > 0; o >>= 1) { if (tid < o) red[tid] += red[tid + o]; __syncthreads(); }
  const float ne = (red[0] > 0.f) ? 1.f : 0.f;
  if (tid == 0) nonempty[be] = ne;
  unsigned short* dst = entcat + (long long)be * 2304;
  for (int d = tid; d < DD; d += 256) {
    float h = row[d], t = trig[b * DD + d];
    dst[d]        = f2bf(h);
    dst[768 + d]  = f2bf(t);
    dst[1536 + d] = f2bf(h * t);
  }
}

// =====================================================================
// L1 normalize score (B,E,NA) over a (t2a) and over e (a2tT, transposed)
// grid B, block 256
// =====================================================================
__global__ void k_l1norm(const float* __restrict__ score, unsigned short* t2a, unsigned short* a2tT)
{
  const int b = blockIdx.x, tid = threadIdx.x;
  __shared__ float s[EE * NA_];
  __shared__ float rs[EE], cs[NA_];
  for (int i = tid; i < EE * NA_; i += 256) s[i] = score[(long long)b * EE * NA_ + i];
  __syncthreads();
  if (tid < EE) {
    float x = 0.f;
    for (int a = 0; a < NA_; ++a) x += fabsf(s[tid * NA_ + a]);
    rs[tid] = fmaxf(x, EPS_);
  } else if (tid < EE + NA_) {
    int a = tid - EE; float x = 0.f;
    for (int e = 0; e < EE; ++e) x += fabsf(s[e * NA_ + a]);
    cs[a] = fmaxf(x, EPS_);
  }
  __syncthreads();
  for (int i = tid; i < EE * NA_; i += 256) {
    int e = i >> 5, a = i & 31;
    t2a[(long long)b * EE * NA_ + i]           = f2bf(s[i] / rs[e]);
    a2tT[(long long)b * EE * NA_ + a * EE + e] = f2bf(s[i] / cs[a]);
  }
}

// ===================== softmax over a2a logits (B,NA,NA) ==============
__global__ void k_softmax(const float* __restrict__ lg, unsigned short* a2ab)
{
  const int b = blockIdx.x, a = threadIdx.x;  // 32 threads
  const float* row = lg + ((long long)b * NA_ + a) * NA_;
  float mx = -1e30f;
  for (int j = 0; j < NA_; ++j) mx = fmaxf(mx, row[j]);
  float sum = 0.f;
  for (int j = 0; j < NA_; ++j) sum += expf(row[j] - mx);
  unsigned short* dst = a2ab + ((long long)b * NA_ + a) * NA_;
  for (int j = 0; j < NA_; ++j) dst[j] = f2bf(expf(row[j] - mx) / sum);
}

// =====================================================================
// t2t: per (b,i): for 3 attentions: gather head-mean at (sent_idx[i],sent_idx[j]),
// mask, row-L1-normalize, average.  Output bf16 [B][SP][KS].
// grid (SP, B), block 128
// =====================================================================
__global__ void k_att(const float* __restrict__ a0, const float* __restrict__ a1,
                      const float* __restrict__ a2, const int* __restrict__ idxs,
                      unsigned short* t2t)
{
  const int i = blockIdx.x, b = blockIdx.y, j = threadIdx.x;  // j in [0,128)
  int nv = 0;
  for (int w = 0; w < WIDX; ++w) nv += (idxs[b * WIDX + w] > 0) ? 1 : 0;
  const bool vi = (i < nv - 2) && (i < SS);
  const bool vj = (j < nv - 2) && (j < SS);
  const int ri = vi ? idxs[b * WIDX + i] : 0;
  const int rj = vj ? idxs[b * WIDX + j] : 0;
  const float* atts[3] = { a0, a1, a2 };
  __shared__ float red[128];
  float acc = 0.f;
  for (int t = 0; t < 3; ++t) {
    float c = 0.f;
    if (vi && vj) {
      const float* base = atts[t] + (long long)b * HH * LL * LL + (long long)ri * LL + rj;
      float s = 0.f;
      for (int h = 0; h < HH; ++h) s += base[(long long)h * LL * LL];
      c = s * (1.f / HH);
    }
    red[j] = c; __syncthreads();
    for (int o = 64; o > 0; o >>= 1) { if (j < o) red[j] += red[j + o]; __syncthreads(); }
    const float rsum = red[0]; __syncthreads();
    acc += c / fmaxf(rsum, EPS_);
  }
  t2t[((long long)b * SP + i) * KS + j] = f2bf(acc * (1.f / 3.f));
}

// =====================================================================
// out[b,e,a] = b2 + sum_n gelu(P_E[b,e,n] + P_A[b,a,n] + b1[n]) * W2[n]
// one wave per output; 8 waves per block
// =====================================================================
__global__ void k_final(const float* __restrict__ PE, const float* __restrict__ PA,
                        const float* __restrict__ b1, const float* __restrict__ W2,
                        const float* __restrict__ b2, float* __restrict__ out)
{
  const int gid  = blockIdx.x * 8 + (threadIdx.x >> 5);
  const int lane = threadIdx.x & 31;
  if (gid >= BB * EE * NA_) return;
  const int b = gid / (EE * NA_);
  const int rem = gid % (EE * NA_);
  const int e = rem / NA_, a = rem % NA_;
  const float* pe = PE + ((long long)b * EE + e) * NHID;
  const float* pa = PA + ((long long)b * NA_ + a) * NHID;
  float acc = 0.f;
  for (int n = lane; n < NHID; n += 32) {
    float x = pe[n] + pa[n] + b1[n];
    float g = 0.5f * x * (1.f + erff(x * 0.70710678118654752f));
    acc += g * W2[n];
  }
  for (int o = 16; o > 0; o >>= 1) acc += __shfl_down(acc, o, 32);
  if (lane == 0) out[gid] = acc + b2[0];
}

// =====================================================================
// Host: workspace layout + launch sequence
// =====================================================================
static inline size_t alloc_off(size_t& off, size_t bytes) {
  size_t o = off;
  off = (off + bytes + 255) & ~(size_t)255;
  return o;
}

extern "C" void kernel_launch(void* const* d_in, const int* in_sizes, int n_in,
                              void* d_out, int out_size, void* d_ws, size_t ws_size,
                              hipStream_t stream) {
  (void)in_sizes; (void)n_in; (void)out_size; (void)ws_size;
  const float* emb    = (const float*)d_in[0];
  const float* istrig = (const float*)d_in[1];
  const float* argw   = (const float*)d_in[2];
  const float* entmap = (const float*)d_in[3];
  const float* att0   = (const float*)d_in[4];
  const float* att1   = (const float*)d_in[5];
  const float* att2   = (const float*)d_in[6];
  const float* W_ta   = (const float*)d_in[7];
  const float* b_ta   = (const float*)d_in[8];
  const float* W1     = (const float*)d_in[9];
  const float* b1     = (const float*)d_in[10];
  const float* W2     = (const float*)d_in[11];
  const float* b2     = (const float*)d_in[12];
  const int*   idxs   = (const int*)d_in[13];
  const int*   blens  = (const int*)d_in[14];
  float* out = (float*)d_out;
  char* ws = (char*)d_ws;

  size_t off = 0;
  unsigned short* sentT  = (unsigned short*)(ws + alloc_off(off, (size_t)BB * DD * KS * 2));
  unsigned short* EMt    = (unsigned short*)(ws + alloc_off(off, (size_t)BB * EE * KS * 2));
  unsigned short* argT   = (unsigned short*)(ws + alloc_off(off, (size_t)BB * DD * KA * 2));
  unsigned short* argWt  = (unsigned short*)(ws + alloc_off(off, (size_t)BB * NA_ * KA * 2));
  float*          trig   = (float*)        (ws + alloc_off(off, (size_t)BB * DD * 4));
  float*          ent0   = (float*)        (ws + alloc_off(off, (size_t)BB * EE * DD * 4));
  unsigned short* entcat = (unsigned short*)(ws + alloc_off(off, (size_t)BB * EE * 2304 * 2));
  float*          nonemp = (float*)        (ws + alloc_off(off, (size_t)BB * EE * 4));
  unsigned short* WtaT   = (unsigned short*)(ws + alloc_off(off, (size_t)DD * 2304 * 2));
  unsigned short* W1ET   = (unsigned short*)(ws + alloc_off(off, (size_t)NHID * 2304 * 2));
  unsigned short* W1AT   = (unsigned short*)(ws + alloc_off(off, (size_t)NHID * 2304 * 2));
  unsigned short* cat3E  = (unsigned short*)(ws + alloc_off(off, (size_t)BB * EE * 2304 * 2));
  unsigned short* cat3A  = (unsigned short*)(ws + alloc_off(off, (size_t)BB * NA_ * 2304 * 2));
  unsigned short* entTb  = (unsigned short*)(ws + alloc_off(off, (size_t)BB * EE * DD * 2));
  unsigned short* entTt  = (unsigned short*)(ws + alloc_off(off, (size_t)BB * DD * EE * 2));
  unsigned short* argTb  = (unsigned short*)(ws + alloc_off(off, (size_t)BB * DD * NA_ * 2));
  float*          score  = (float*)        (ws + alloc_off(off, (size_t)BB * EE * NA_ * 4));
  unsigned short* t2a    = (unsigned short*)(ws + alloc_off(off, (size_t)BB * EE * NA_ * 2));
  unsigned short* a2tT   = (unsigned short*)(ws + alloc_off(off, (size_t)BB * NA_ * EE * 2));
  float*          a2aL   = (float*)        (ws + alloc_off(off, (size_t)BB * NA_ * NA_ * 4));
  unsigned short* a2ab   = (unsigned short*)(ws + alloc_off(off, (size_t)BB * NA_ * NA_ * 2));
  unsigned short* t2t    = (unsigned short*)(ws + alloc_off(off, (size_t)BB * SP * KS * 2));
  unsigned short* ahhT   = (unsigned short*)(ws + alloc_off(off, (size_t)BB * DD * KS * 2));
  float*          PE     = (float*)        (ws + alloc_off(off, (size_t)BB * EE * NHID * 4));
  float*          PA     = (float*)        (ws + alloc_off(off, (size_t)BB * NA_ * NHID * 4));

  // Only ahhT has K-padding (cols 112..127) that is never written; every other
  // buffer is fully overwritten each call.
  hipMemsetAsync(ahhT, 0, (size_t)BB * DD * KS * 2, stream);

  // weight packing
  {
    long long tot = 768LL * 2304 + 2LL * 1024 * 2304;
    int blocks = (int)((tot + 255) / 256);
    k_pack_w<<<blocks, 256, 0, stream>>>(W_ta, W1, WtaT, W1ET, W1AT);
  }

  // gathers / transposes / trigger mean
  k_gather<<<dim3(BB, 4), 256, 0, stream>>>(emb, istrig, argw, entmap, idxs, blens,
                                            sentT, EMt, argT, argWt, trig);

  const float rsqD = 1.0f / sqrtf((float)DD);

  #define GEMM(BM_, BN_, A_, sA_, ldA_, B_, sB_, ldB_, Cf_, sCf_, ldCf_,           \
               Cb_, sCb_, ldCb_, Ct_, sCt_, ldCt_, bias_, mask_, sM_, sc_,         \
               Mt_, Nt_, K_)                                                       \
    k_gemm_bf16<BM_, BN_><<<dim3((Nt_) / (BN_), (Mt_) / (BM_), BB), 32, 0, stream>>>( \
        (const unsigned short*)(A_), (long long)(sA_), (ldA_),                     \
        (const unsigned short*)(B_), (long long)(sB_), (ldB_),                     \
        (float*)(Cf_), (long long)(sCf_), (ldCf_),                                 \
        (unsigned short*)(Cb_), (long long)(sCb_), (ldCb_),                        \
        (unsigned short*)(Ct_), (long long)(sCt_), (ldCt_),                        \
        (bias_), (mask_), (long long)(sM_), (sc_), (K_))

  // G1: entity0/H_ (E x D) = EMt (E x KS) * sentT^T ; bf16 copy -> cat3E col 0
  GEMM(2, 4, EMt, EE * KS, KS, sentT, DD * KS, KS,
       ent0, EE * DD, DD, cat3E, EE * 2304, 2304,
       nullptr, 0, 0, nullptr, nullptr, 0, 1.0f, EE / 16, DD / 16, KS);

  // G2: arg (NA x D) = argWt (NA x KA) * argT^T ; bf16 -> cat3A col 0 ; Ct -> argTb [D x NA]
  GEMM(2, 4, argWt, NA_ * KA, KA, argT, DD * KA, KA,
       nullptr, 0, 0, cat3A, NA_ * 2304, 2304,
       argTb, DD * NA_, NA_, nullptr, nullptr, 0, 1.0f, NA_ / 16, DD / 16, KA);

  // ent_cat + nonempty
  k_entcat<<<BB * EE, 256, 0, stream>>>(ent0, trig, entcat, nonemp);

  // G3: entity_ta (E x D) = entcat (E x 2304) * WtaT^T, +b_ta, *nonempty
  GEMM(2, 4, entcat, EE * 2304, 2304, WtaT, 0, 2304,
       nullptr, 0, 0, entTb, EE * DD, DD,
       entTt, DD * EE, EE, b_ta, nonemp, EE, 1.0f, EE / 16, DD / 16, 2304);

  // G4: score (E x NA) = entTb (E x D) * arg^T (cat3A col0), / sqrt(D)
  GEMM(2, 2, entTb, EE * DD, DD, cat3A, NA_ * 2304, 2304,
       score, EE * NA_, NA_, nullptr, 0, 0,
       nullptr, 0, 0, nullptr, nullptr, 0, rsqD, EE / 16, NA_ / 16, DD);

  k_l1norm<<<BB, 256, 0, stream>>>(score, t2a, a2tT);

  // G5: tokArgAwared (E x D) = t2a (E x NA) * argTb^T -> cat3E col 768
  GEMM(2, 4, t2a, EE * NA_, NA_, argTb, DD * NA_, NA_,
       nullptr, 0, 0, cat3E + 768, EE * 2304, 2304,
       nullptr, 0, 0, nullptr, nullptr, 0, 1.0f, EE / 16, DD / 16, NA_);

  // G6: argTokAwared (NA x D) = a2tT (NA x E) * entTt^T -> cat3A col 768
  GEMM(2, 4, a2tT, NA_ * EE, EE, entTt, DD * EE, EE,
       nullptr, 0, 0, cat3A + 768, NA_ * 2304, 2304,
       nullptr, 0, 0, nullptr, nullptr, 0, 1.0f, NA_ / 16, DD / 16, EE);

  // G7: a2a logits (NA x NA) = arg * arg^T
  GEMM(2, 2, cat3A, NA_ * 2304, 2304, cat3A, NA_ * 2304, 2304,
       a2aL, NA_ * NA_, NA_, nullptr, 0, 0,
       nullptr, 0, 0, nullptr, nullptr, 0, 1.0f, NA_ / 16, NA_ / 16, DD);

  k_softmax<<<BB, 32, 0, stream>>>(a2aL, a2ab);

  // G8: A_u2u (NA x D) = a2ab (NA x NA) * argTb^T -> cat3A col 1536
  GEMM(2, 4, a2ab, NA_ * NA_, NA_, argTb, DD * NA_, NA_,
       nullptr, 0, 0, cat3A + 1536, NA_ * 2304, 2304,
       nullptr, 0, 0, nullptr, nullptr, 0, 1.0f, NA_ / 16, DD / 16, NA_);

  // attention head-mean gather + row-normalize + average
  k_att<<<dim3(SP, BB), 128, 0, stream>>>(att0, att1, att2, idxs, t2t);

  // G9: a_h2h_tok (SP x D) = t2t (SP x KS) * sentT^T ; only Ct -> ahhT [D x KS]
  GEMM(1, 4, t2t, SP * KS, KS, sentT, DD * KS, KS,
       nullptr, 0, 0, nullptr, 0, 0,
       ahhT, DD * KS, KS, nullptr, nullptr, 0, 1.0f, SP / 16, DD / 16, KS);

  // G10: A_h2h (E x D) = EMt (E x KS) * ahhT^T -> cat3E col 1536
  GEMM(2, 4, EMt, EE * KS, KS, ahhT, DD * KS, KS,
       nullptr, 0, 0, cat3E + 1536, EE * 2304, 2304,
       nullptr, 0, 0, nullptr, nullptr, 0, 1.0f, EE / 16, DD / 16, KS);

  // G11: P_E (E x NHID) = cat3E (E x 2304) * W1ET^T
  GEMM(2, 4, cat3E, EE * 2304, 2304, W1ET, 0, 2304,
       PE, EE * NHID, NHID, nullptr, 0, 0,
       nullptr, 0, 0, nullptr, nullptr, 0, 1.0f, EE / 16, NHID / 16, 2304);

  // G12: P_A (NA x NHID) = cat3A (NA x 2304) * W1AT^T
  GEMM(2, 4, cat3A, NA_ * 2304, 2304, W1AT, 0, 2304,
       PA, NA_ * NHID, NHID, nullptr, 0, 0,
       nullptr, 0, 0, nullptr, nullptr, 0, 1.0f, NA_ / 16, NHID / 16, 2304);

  // final GELU-MLP contraction against W2
  k_final<<<(BB * EE * NA_) / 8, 256, 0, stream>>>(PE, PA, b1, W2, b2, out);

  #undef GEMM
}
